// AGRACE_87144886436441
// MI455X (gfx1250) — matrix operand order repaired and checked by reading, not compiled, not run on hardware
//
#include <hip/hip_runtime.h>
#include <climits>

// Problem dims (match reference)
#define BDIM  4
#define SDIM  2048
#define DIN   4096
#define DOUT  4096
#define ENCD  256
#define KKEYS 2048
#define MTOT  (BDIM * SDIM)   // 8192

typedef __attribute__((ext_vector_type(16))) __bf16 v16bf;
typedef __attribute__((ext_vector_type(8)))  float  v8f;
typedef __attribute__((ext_vector_type(4)))  int    v4i_a;

union Frag32B { uint4 u[2]; v16bf v; };

// round-to-nearest-even f32 -> bf16, two at a time packed into one dword
__device__ __forceinline__ unsigned pack2bf(float lo, float hi) {
    unsigned a = __builtin_bit_cast(unsigned, lo);
    unsigned b = __builtin_bit_cast(unsigned, hi);
    a += 0x7fffu + ((a >> 16) & 1u);
    b += 0x7fffu + ((b >> 16) & 1u);
    return (a >> 16) | (b & 0xffff0000u);
}

// ---------------- CDNA5 async global->LDS helpers (ASYNCcnt path) ----------
__device__ __forceinline__ void async_copy_b128(unsigned ldsOff, const void* g) {
#if __has_builtin(__builtin_amdgcn_global_load_async_to_lds_b128)
    __builtin_amdgcn_global_load_async_to_lds_b128(
        (__attribute__((address_space(1))) v4i_a*)(uintptr_t)g,
        (__attribute__((address_space(3))) v4i_a*)(uintptr_t)ldsOff,
        0, 0);
#else
    unsigned long long ga = (unsigned long long)(uintptr_t)g;
    asm volatile("global_load_async_to_lds_b128 %0, %1, off"
                 :: "v"(ldsOff), "v"(ga) : "memory");
#endif
}

__device__ __forceinline__ void wait_async0() {
#if __has_builtin(__builtin_amdgcn_s_wait_asynccnt)
    __builtin_amdgcn_s_wait_asynccnt(0);
#else
    asm volatile("s_wait_asynccnt 0x0" ::: "memory");
#endif
}

// ---------------------------------------------------------------- kernel 1a
__global__ void ep_init_kernel(int* first_raw) {
    if (threadIdx.x < BDIM) first_raw[threadIdx.x] = INT_MAX;
}

// ---------------------------------------------------------------- kernel 1b
__global__ __launch_bounds__(256) void ep_diff_kernel(const float* __restrict__ x,
                                                      int* __restrict__ first_raw) {
    const int b = blockIdx.x / (SDIM - 1);
    const int j = blockIdx.x % (SDIM - 1) + 1;
    const float* r0 = x + ((size_t)b * SDIM + (j - 1)) * DIN;
    const float* r1 = r0 + DIN;
    bool diff = false;
    for (int d = threadIdx.x; d < DIN; d += 256)
        diff |= (r0[d] != r1[d]);
    if (diff) atomicMin(&first_raw[b], j);
}

// ---------------------------------------------------------------- kernel 1c
__global__ void ep_finalize_kernel(const int* __restrict__ first_raw,
                                   int* __restrict__ first) {
    if (threadIdx.x < BDIM) {
        int v = first_raw[threadIdx.x];
        first[threadIdx.x] = (v == INT_MAX || v == 1) ? 0 : v;
    }
}

// ---------------------------------------------------------------- kernel 2
__global__ __launch_bounds__(256) void ep_pool_kernel(const float* __restrict__ x,
                                                      const int* __restrict__ first,
                                                      float* __restrict__ pooled) {
    const int idx = blockIdx.x * 256 + threadIdx.x;   // over BDIM*DIN
    const int b = idx >> 12;          // DIN == 4096
    const int d = idx & (DIN - 1);
    const int f = first[b];
    const float* p = x + ((size_t)b * SDIM + f) * DIN + d;
    float s = 0.f;
    for (int j = f; j < SDIM; ++j) { s += *p; p += DIN; }
    pooled[b * DIN + d] = s / (float)(SDIM - f);
}

// ---------------------------------------------------------------- kernel 3
__global__ __launch_bounds__(256) void ep_encoder_kernel(
        const float* __restrict__ pooled,
        const float* __restrict__ W1, const float* __restrict__ b1,
        const float* __restrict__ W2, const float* __restrict__ b2,
        const float* __restrict__ keys, const float* __restrict__ eps,
        int* __restrict__ chosen, int* __restrict__ hitflag) {
    __shared__ float sh[BDIM][ENCD];
    __shared__ float sq[BDIM][ENCD];
    __shared__ float sd[256];
    __shared__ int   si[256];
    const int t = threadIdx.x;

    for (int b = 0; b < BDIM; ++b) {
        const float* pr = pooled + b * DIN;
        const float* wr = W1 + (size_t)t * DIN;
        float s = b1[t];
        for (int d = 0; d < DIN; ++d) s += pr[d] * wr[d];
        sh[b][t] = fmaxf(s, 0.f);
    }
    __syncthreads();
    for (int b = 0; b < BDIM; ++b) {
        const float* wr = W2 + (size_t)t * ENCD;
        float s = b2[t];
        for (int j = 0; j < ENCD; ++j) s += sh[b][j] * wr[j];
        sq[b][t] = s;
    }
    __syncthreads();
    for (int b = 0; b < BDIM; ++b) {
        float best = 3.4e38f;
        int bestk = INT_MAX;
        for (int k = t; k < KKEYS; k += 256) {
            const float* kr = keys + (size_t)k * ENCD;
            float d2 = 0.f;
            for (int e = 0; e < ENCD; ++e) {
                float df = kr[e] - sq[b][e];
                d2 += df * df;
            }
            float dist = sqrtf(fmaxf(d2, 0.f));
            if (dist < best || (dist == best && k < bestk)) { best = dist; bestk = k; }
        }
        sd[t] = best; si[t] = bestk;
        __syncthreads();
        for (int s = 128; s > 0; s >>= 1) {
            if (t < s) {
                if (sd[t + s] < sd[t] || (sd[t + s] == sd[t] && si[t + s] < si[t])) {
                    sd[t] = sd[t + s]; si[t] = si[t + s];
                }
            }
            __syncthreads();
        }
        if (t == 0) {
            int k = si[0];
            chosen[b]  = k;
            hitflag[b] = (sd[0] <= eps[k]) ? 1 : 0;
        }
        __syncthreads();
    }
}

// ---------------------------------------------------------------- kernel 4a
// bulk f32 -> bf16 conversion (8 elements / thread)
__global__ __launch_bounds__(256) void ep_cvt_kernel(const float* __restrict__ src,
                                                     uint4* __restrict__ dst) {
    const size_t i = (size_t)blockIdx.x * 256 + threadIdx.x;
    const float4* s = (const float4*)src + 2 * i;
    float4 a = s[0], b = s[1];
    uint4 p;
    p.x = pack2bf(a.x, a.y); p.y = pack2bf(a.z, a.w);
    p.z = pack2bf(b.x, b.y); p.w = pack2bf(b.z, b.w);
    dst[i] = p;
}

// ---------------------------------------------------------------- kernel 4b
// bf16 WMMA GEMM with async global->LDS double buffering.
// block tile 256(M)x128(N), 8 waves (4 along M x 2 along N), wave tile 64x64,
// K-step 32, LDS rows padded to 80B (bank-conflict-free b128 reads).
#define LSTRIDE 80
#define AT_ROWS 256
#define BT_ROWS 128
#define BUF_BYTES ((AT_ROWS + BT_ROWS) * LSTRIDE)   // 30720

__device__ __forceinline__ void ep_issue_tiles(const __bf16* __restrict__ xbf,
                                               const __bf16* __restrict__ wbf,
                                               int m0, int n0, int k0,
                                               unsigned ldsA, unsigned ldsB, int tid) {
    #pragma unroll
    for (int r = 0; r < 4; ++r) {              // A: 256 rows x 64B = 1024 x b128
        int i = tid + 256 * r;
        int row = i >> 2, q = i & 3;
        async_copy_b128(ldsA + row * LSTRIDE + q * 16,
                        (const char*)(xbf + (size_t)(m0 + row) * DIN + k0) + q * 16);
    }
    #pragma unroll
    for (int r = 0; r < 2; ++r) {              // B: 128 rows x 64B = 512 x b128
        int i = tid + 256 * r;
        int row = i >> 2, q = i & 3;
        async_copy_b128(ldsB + row * LSTRIDE + q * 16,
                        (const char*)(wbf + (size_t)(n0 + row) * DIN + k0) + q * 16);
    }
}

__global__ __launch_bounds__(256) void ep_gemm_bf16_kernel(
        const __bf16* __restrict__ xbf, const __bf16* __restrict__ wbf,
        const float* __restrict__ bias, const float* __restrict__ values,
        const int* __restrict__ chosen, const int* __restrict__ hitflag,
        float* __restrict__ out) {
    __shared__ __attribute__((aligned(16))) char lds[2 * BUF_BYTES];

    const int tid = threadIdx.x;
    const int m0  = blockIdx.x * 256;
    const int n0  = blockIdx.y * 128;
    const int bidx = m0 >> 11;                 // 256 | 2048: tile within one sample

    if (hitflag[bidx]) {                       // whole sample replaced
        const float* vrow = values + (size_t)chosen[bidx] * DOUT + n0;
        for (int idx = tid; idx < 256 * 128; idx += 256) {
            int rr = idx >> 7, cc = idx & 127;
            out[(size_t)(m0 + rr) * DOUT + n0 + cc] = vrow[cc];
        }
        return;
    }

    const int wave = tid >> 5;
    const int lane = tid & 31;
    const int wm = wave & 3;                   // M group 0..3 (64 rows each)
    const int wn = wave >> 2;                  // N group 0..1 (64 cols each)
    const int rsel = lane & 15;
    const bool hiHalf = (lane >= 16);
    const int aOff = hiHalf ? 16 : 0;          // A frag: K {0..7,16..23} vs {8..15,24..31}
    const int bOff = hiHalf ? 16 : 0;          // B frag: K {0..7,16..23} vs {8..15,24..31}

    const unsigned ldsBase = (unsigned)(uintptr_t)(void*)lds;

    v8f acc[4][4] = {};

    // prologue: DMA first tile pair into buffer 0
    ep_issue_tiles(xbf, wbf, m0, n0, 0, ldsBase, ldsBase + AT_ROWS * LSTRIDE, tid);

    for (int k0 = 0; k0 < DIN; k0 += 32) {
        const int cur = (k0 >> 5) & 1;
        const unsigned bufOff = cur * BUF_BYTES;

        wait_async0();                         // this wave's current tiles landed
        __syncthreads();                       // all waves' tiles visible; prior reads done

        if (k0 + 32 < DIN) {                   // DMA next tiles, overlapped with WMMA
            const unsigned nb = (1 - cur) * BUF_BYTES;
            ep_issue_tiles(xbf, wbf, m0, n0, k0 + 32,
                           ldsBase + nb, ldsBase + nb + AT_ROWS * LSTRIDE, tid);
        }

        const char* bufA = lds + bufOff;
        const char* bufB = lds + bufOff + AT_ROWS * LSTRIDE;

        Frag32B af[4];
        #pragma unroll
        for (int mi = 0; mi < 4; ++mi) {
            const char* pr = bufA + (wm * 64 + mi * 16 + rsel) * LSTRIDE;
            af[mi].u[0] = *(const uint4*)(pr + aOff);
            af[mi].u[1] = *(const uint4*)(pr + aOff + 32);
        }
        Frag32B bfg[4];
        #pragma unroll
        for (int ni = 0; ni < 4; ++ni) {
            const char* pr = bufB + (wn * 64 + ni * 16 + rsel) * LSTRIDE;
            bfg[ni].u[0] = *(const uint4*)(pr + bOff);
            bfg[ni].u[1] = *(const uint4*)(pr + bOff + 32);
        }
        #pragma unroll
        for (int mi = 0; mi < 4; ++mi)
            #pragma unroll
            for (int ni = 0; ni < 4; ++ni)
                acc[mi][ni] = __builtin_amdgcn_wmma_f32_16x16x32_bf16(
                    false, af[mi].v, false, bfg[ni].v,
                    (short)0, acc[mi][ni], false, false);
    }

    #pragma unroll
    for (int ni = 0; ni < 4; ++ni) {
        const int n = n0 + wn * 64 + ni * 16 + rsel;
        const float bv = bias[n];
        #pragma unroll
        for (int mi = 0; mi < 4; ++mi) {
            const int mbase = m0 + wm * 64 + mi * 16 + (hiHalf ? 8 : 0);
            #pragma unroll
            for (int i = 0; i < 8; ++i)
                out[(size_t)(mbase + i) * DOUT + n] = acc[mi][ni][i] + bv;
        }
    }
}

// ---------------------------------------------------------------- kernel 4c
// fallback: f32-load + in-loop bf16 convert GEMM (used when ws too small
// for the bf16 mirrors). block 128x128, wave 32x64.
#define LDS_STRIDE 80
__global__ __launch_bounds__(256) void ep_gemm_kernel(
        const float* __restrict__ x, const float* __restrict__ W,
        const float* __restrict__ bias, const float* __restrict__ values,
        const int* __restrict__ chosen, const int* __restrict__ hitflag,
        float* __restrict__ out) {
    __shared__ __attribute__((aligned(16))) char lds[2 * 128 * LDS_STRIDE];
    char* ldsA = lds;
    char* ldsB = lds + 128 * LDS_STRIDE;

    const int tid = threadIdx.x;
    const int m0  = blockIdx.x * 128;
    const int n0  = blockIdx.y * 128;
    const int bidx = m0 >> 11;

    if (hitflag[bidx]) {
        const float* vrow = values + (size_t)chosen[bidx] * DOUT + n0;
        for (int idx = tid; idx < 128 * 128; idx += 256) {
            int rr = idx >> 7, cc = idx & 127;
            out[(size_t)(m0 + rr) * DOUT + n0 + cc] = vrow[cc];
        }
        return;
    }

    const int wave = tid >> 5;
    const int lane = tid & 31;
    const int wm = wave & 3;
    const int wn = wave >> 2;
    const int rsel = lane & 15;
    const bool hiHalf = (lane >= 16);

    const int ldr = tid >> 1;
    const int ldh = tid & 1;

    v8f acc[2][4] = {};

    const float* gA = x + (size_t)(m0 + ldr) * DIN + ldh * 16;
    const float* gB = W + (size_t)(n0 + ldr) * DIN + ldh * 16;
    char* stA = ldsA + ldr * LDS_STRIDE + ldh * 32;
    char* stB = ldsB + ldr * LDS_STRIDE + ldh * 32;

    const int aOff = hiHalf ? 16 : 0;
    const int bOff = hiHalf ? 16 : 0;

    for (int k0 = 0; k0 < DIN; k0 += 32) {
        const float4* pa = (const float4*)(gA + k0);
        const float4* pb = (const float4*)(gB + k0);
        float4 a0 = pa[0], a1 = pa[1], a2 = pa[2], a3 = pa[3];
        float4 b0 = pb[0], b1 = pb[1], b2 = pb[2], b3 = pb[3];
        if (k0 + 32 < DIN) {
            __builtin_prefetch(gA + k0 + 32, 0, 1);
            __builtin_prefetch(gB + k0 + 32, 0, 1);
        }
        uint4 pA0, pA1, pB0, pB1;
        pA0.x = pack2bf(a0.x, a0.y); pA0.y = pack2bf(a0.z, a0.w);
        pA0.z = pack2bf(a1.x, a1.y); pA0.w = pack2bf(a1.z, a1.w);
        pA1.x = pack2bf(a2.x, a2.y); pA1.y = pack2bf(a2.z, a2.w);
        pA1.z = pack2bf(a3.x, a3.y); pA1.w = pack2bf(a3.z, a3.w);
        pB0.x = pack2bf(b0.x, b0.y); pB0.y = pack2bf(b0.z, b0.w);
        pB0.z = pack2bf(b1.x, b1.y); pB0.w = pack2bf(b1.z, b1.w);
        pB1.x = pack2bf(b2.x, b2.y); pB1.y = pack2bf(b2.z, b2.w);
        pB1.z = pack2bf(b3.x, b3.y); pB1.w = pack2bf(b3.z, b3.w);

        __syncthreads();
        *(uint4*)(stA)      = pA0;
        *(uint4*)(stA + 16) = pA1;
        *(uint4*)(stB)      = pB0;
        *(uint4*)(stB + 16) = pB1;
        __syncthreads();

        Frag32B af[2];
        #pragma unroll
        for (int mi = 0; mi < 2; ++mi) {
            const char* pr = ldsA + (wm * 32 + mi * 16 + rsel) * LDS_STRIDE;
            af[mi].u[0] = *(const uint4*)(pr + aOff);
            af[mi].u[1] = *(const uint4*)(pr + aOff + 32);
        }
        Frag32B bfg[4];
        #pragma unroll
        for (int ni = 0; ni < 4; ++ni) {
            const char* pr = ldsB + (wn * 64 + ni * 16 + rsel) * LDS_STRIDE;
            bfg[ni].u[0] = *(const uint4*)(pr + bOff);
            bfg[ni].u[1] = *(const uint4*)(pr + bOff + 32);
        }
        #pragma unroll
        for (int mi = 0; mi < 2; ++mi)
            #pragma unroll
            for (int ni = 0; ni < 4; ++ni)
                acc[mi][ni] = __builtin_amdgcn_wmma_f32_16x16x32_bf16(
                    false, af[mi].v, false, bfg[ni].v,
                    (short)0, acc[mi][ni], false, false);
    }

    #pragma unroll
    for (int ni = 0; ni < 4; ++ni) {
        const int n = n0 + wn * 64 + ni * 16 + rsel;
        const float bv = bias[n];
        #pragma unroll
        for (int mi = 0; mi < 2; ++mi) {
            const int mbase = m0 + wm * 32 + mi * 16 + (hiHalf ? 8 : 0);
            #pragma unroll
            for (int i = 0; i < 8; ++i)
                out[(size_t)(mbase + i) * DOUT + n] = acc[mi][ni][i] + bv;
        }
    }
}

// ---------------------------------------------------------------- launch
extern "C" void kernel_launch(void* const* d_in, const int* in_sizes, int n_in,
                              void* d_out, int out_size, void* d_ws, size_t ws_size,
                              hipStream_t stream) {
    const float* x      = (const float*)d_in[0];
    const float* W      = (const float*)d_in[1];
    const float* bias   = (const float*)d_in[2];
    const float* enc_W1 = (const float*)d_in[3];
    const float* enc_b1 = (const float*)d_in[4];
    const float* enc_W2 = (const float*)d_in[5];
    const float* enc_b2 = (const float*)d_in[6];
    const float* keys   = (const float*)d_in[7];
    const float* values = (const float*)d_in[8];
    const float* eps    = (const float*)d_in[9];
    float* out = (float*)d_out;

    // workspace layout
    int*   first_raw = (int*)d_ws;                       // [4]
    int*   first     = first_raw + 4;                    // [4]
    int*   chosen    = first_raw + 8;                    // [4]
    int*   hitflag   = first_raw + 12;                   // [4]
    float* pooled    = (float*)(first_raw + 16);         // [4*4096]

    const size_t XB_OFF   = 65792;                       // 256-aligned, past pooled
    const size_t XB_BYTES = (size_t)MTOT * DIN * 2;      // 64 MB bf16 x
    const size_t WB_BYTES = (size_t)DOUT * DIN * 2;      // 32 MB bf16 W
    const size_t NEED     = XB_OFF + XB_BYTES + WB_BYTES;
    const bool   fast     = (ws_size >= NEED);

    __bf16* xbf = (__bf16*)((char*)d_ws + XB_OFF);
    __bf16* wbf = (__bf16*)((char*)d_ws + XB_OFF + XB_BYTES);

    ep_init_kernel<<<1, 32, 0, stream>>>(first_raw);
    ep_diff_kernel<<<BDIM * (SDIM - 1), 256, 0, stream>>>(x, first_raw);
    ep_finalize_kernel<<<1, 32, 0, stream>>>(first_raw, first);
    ep_pool_kernel<<<(BDIM * DIN) / 256, 256, 0, stream>>>(x, first, pooled);
    ep_encoder_kernel<<<1, 256, 0, stream>>>(pooled, enc_W1, enc_b1, enc_W2, enc_b2,
                                             keys, eps, chosen, hitflag);
    if (fast) {
        ep_cvt_kernel<<<(MTOT * (size_t)DIN / 8) / 256, 256, 0, stream>>>(x, (uint4*)xbf);
        ep_cvt_kernel<<<((size_t)DOUT * DIN / 8) / 256, 256, 0, stream>>>(W, (uint4*)wbf);
        dim3 grid(MTOT / 256, DOUT / 128);
        ep_gemm_bf16_kernel<<<grid, 256, 0, stream>>>(xbf, wbf, bias, values,
                                                      chosen, hitflag, out);
    } else {
        dim3 grid(MTOT / 128, DOUT / 128);
        ep_gemm_kernel<<<grid, 256, 0, stream>>>(x, W, bias, values,
                                                 chosen, hitflag, out);
    }
}
